// ChildSumTreeLSTMRNNEncoder_79190607003721
// MI455X (gfx1250) — compile-verified, matrix operand order checked by reading, not run on hardware
//
#include <hip/hip_runtime.h>
#include <hip/hip_bf16.h>

// ---------------------------------------------------------------------------
// ChildSum TreeLSTM + GRU encoder for MI455X (gfx1250, wave32, WMMA).
// All matrix math runs through v_wmma_f32_16x16x32_bf16 (f32 accumulate).
// GEMM is templated on <K, LDA, LDC> so the K loop fully unrolls and all
// strides fold into global_load/store immediate offsets (no per-access
// 64-bit address math).
// ---------------------------------------------------------------------------

typedef __attribute__((ext_vector_type(16))) __bf16 bf16x16;
typedef __attribute__((ext_vector_type(8)))  __bf16 bf16x8;
typedef __attribute__((ext_vector_type(8)))  float  f32x8;

// Problem constants (fixed by the reference).
constexpr int N   = 8192;
constexpr int L   = 8;
constexpr int I   = 256;
constexpr int H   = 512;
constexpr int D   = 8;
constexpr int PER = N / D;      // 1024 nodes per depth level
constexpr int G3  = 3 * H;      // 1536
constexpr int G4  = 4 * H;      // 2048
constexpr int TB  = 256;

__device__ __forceinline__ float sigmoidf_(float x) {
  return 1.0f / (1.0f + __expf(-x));
}

// Load one 16-bf16 WMMA fragment for this lane: two contiguous 16B chunks.
// CDNA5 16-bit A/B layout: lanes 0-15 hold K = [k..k+7] then [k+16..k+23],
// lanes 16-31 hold K = [k+8..k+15] then [k+24..k+31]; caller passes p0 at
// (k + (lane>>4)*8) so p1 = p0 + 16.
__device__ __forceinline__ bf16x16 load_frag(const __bf16* p0) {
  bf16x8 lo = *reinterpret_cast<const bf16x8*>(p0);
  bf16x8 hi = *reinterpret_cast<const bf16x8*>(p0 + 16);
  bf16x16 v;
#pragma unroll
  for (int i = 0; i < 8; ++i) { v[i] = lo[i]; v[i + 8] = hi[i]; }
  return v;
}

// C[M,Nc] = A[M,K](bf16,row-major,LDA) * Bt[Nc,K](bf16, pre-transposed) + bias
// Requires M%128==0, Nc%128==0, K%32==0 (true for all shapes here).
// Block = 256 threads = 8 waves; block tile 128x128; wave tile 64x32.
template <int K, int LDA, int LDC>
__global__ __launch_bounds__(256) void k_gemm_bf16(
    const __bf16* __restrict__ A,
    const __bf16* __restrict__ Bt,
    const float* __restrict__ bias,
    float* __restrict__ C) {
  const int lane  = threadIdx.x & 31;
  const int wave  = threadIdx.x >> 5;
  const int waveM = wave >> 2;                    // 0..1
  const int waveN = wave & 3;                     // 0..3
  const int row0  = blockIdx.y * 128 + waveM * 64;
  const int col0  = blockIdx.x * 128 + waveN * 32;
  const int r     = lane & 15;
  const int half  = lane >> 4;

  // Per-lane base pointers; K-step advances become immediate offsets.
  const __bf16* pa0 = A + (size_t)(row0 + r) * LDA + half * 8;
  const __bf16* pb0 = Bt + (size_t)(col0 + r) * K + half * 8;

  f32x8 acc[4][2] = {};

#pragma unroll
  for (int k = 0; k < K; k += 32) {
    bf16x16 af[4];
    bf16x16 bfr[2];
#pragma unroll
    for (int i = 0; i < 4; ++i)
      af[i] = load_frag(pa0 + (size_t)(16 * i) * LDA + k);
#pragma unroll
    for (int j = 0; j < 2; ++j)
      bfr[j] = load_frag(pb0 + (size_t)(16 * j) * K + k);
#pragma unroll
    for (int i = 0; i < 4; ++i)
#pragma unroll
      for (int j = 0; j < 2; ++j)
        acc[i][j] = __builtin_amdgcn_wmma_f32_16x16x32_bf16(
            false, af[i], false, bfr[j], (short)0, acc[i][j], false, false);
  }

  // C/D layout: lane holds column col0+16j+(lane&15); VGPR v holds row
  // row0+16i + v + 8*(lane>>4). All row offsets are compile-time immediates.
#pragma unroll
  for (int j = 0; j < 2; ++j) {
    const int col = col0 + 16 * j + r;
    const float bv = bias ? bias[col] : 0.0f;
    float* Cb = C + (size_t)(row0 + half * 8) * LDC + col;
#pragma unroll
    for (int i = 0; i < 4; ++i)
#pragma unroll
      for (int v = 0; v < 8; ++v)
        Cb[(size_t)(16 * i + v) * LDC] = acc[i][j][v] + bv;
  }
}

// ---------------------------------------------------------------------------
// Elementwise / scatter kernels
// ---------------------------------------------------------------------------

__global__ void k_cvt_bf16(const float* __restrict__ in, __bf16* __restrict__ out, int n) {
  int idx = blockIdx.x * blockDim.x + threadIdx.x;
  if (idx < n) out[idx] = (__bf16)in[idx];
}

// in: [K, Nc] f32 row-major  ->  out: [Nc, K] bf16 (transposed for WMMA B)
__global__ void k_cvt_transpose(const float* __restrict__ in, __bf16* __restrict__ out,
                                int K, int Nc) {
  int idx = blockIdx.x * blockDim.x + threadIdx.x;
  if (idx >= K * Nc) return;
  int k = idx / Nc, n2 = idx % Nc;
  out[(size_t)n2 * K + k] = (__bf16)in[idx];
}

__global__ void k_zero_f32(float* p, int n) {
  int idx = blockIdx.x * blockDim.x + threadIdx.x;
  if (idx < n) p[idx] = 0.0f;
}
__global__ void k_zero_bf16(__bf16* p, int n) {
  int idx = blockIdx.x * blockDim.x + threadIdx.x;
  if (idx < n) p[idx] = (__bf16)0.0f;
}

// GRU gate update. gi includes b_ih, gh includes b_hh (fused in GEMM epilogue).
__global__ void k_gru_gate(const float* __restrict__ gi, const float* __restrict__ gh,
                           float* __restrict__ h, __bf16* __restrict__ hb) {
  int idx = blockIdx.x * blockDim.x + threadIdx.x;
  if (idx >= N * H) return;
  int n = idx / H, d = idx % H;
  const float* gin = gi + (size_t)n * G3;
  const float* ghn = gh + (size_t)n * G3;
  float rg = sigmoidf_(gin[d] + ghn[d]);
  float z  = sigmoidf_(gin[H + d] + ghn[H + d]);
  float ng = tanhf(gin[2 * H + d] + rg * ghn[2 * H + d]);
  float hn = (1.0f - z) * ng + z * h[idx];
  h[idx]  = hn;
  hb[idx] = (__bf16)hn;
}

// h_sum[parent - pbase] += h[child], children = rows [cbase, cbase+PER)
__global__ void k_hsum_scatter(const float* __restrict__ ht, const int* __restrict__ parent,
                               int cbase, int pbase, float* __restrict__ hsum) {
  int idx = blockIdx.x * blockDim.x + threadIdx.x;
  if (idx >= PER * H) return;
  int j = cbase + idx / H, d = idx % H;
  int p = parent[j];
  atomicAdd(&hsum[(size_t)(p - pbase) * H + d], ht[(size_t)j * H + d]);
}

// fc[parent - pbase] += sigmoid(xf[parent] + (h_child @ U_f)) * c[child]
__global__ void k_fc_scatter(const float* __restrict__ xg, const float* __restrict__ hUf,
                             const float* __restrict__ c, const int* __restrict__ parent,
                             int cbase, int pbase, float* __restrict__ fc) {
  int idx = blockIdx.x * blockDim.x + threadIdx.x;
  if (idx >= PER * H) return;
  int jj = idx / H, d = idx % H;
  int j = cbase + jj;
  int p = parent[j];
  float f = sigmoidf_(xg[(size_t)p * G4 + 3 * H + d] + hUf[(size_t)jj * H + d]);
  atomicAdd(&fc[(size_t)(p - pbase) * H + d], f * c[(size_t)j * H + d]);
}

// c_new = sigmoid(xi+gi)*tanh(xu+gu) + fc ; h = sigmoid(xo+go)*tanh(c_new)
__global__ void k_level_update(const float* __restrict__ xg, const float* __restrict__ giou,
                               const float* __restrict__ fc, int pbase,
                               float* __restrict__ c, float* __restrict__ ht,
                               __bf16* __restrict__ htb) {
  int idx = blockIdx.x * blockDim.x + threadIdx.x;
  if (idx >= PER * H) return;
  int rr = idx / H, d = idx % H;
  int rw = pbase + rr;
  const float* x = xg + (size_t)rw * G4;
  const float* g = giou + (size_t)rr * G3;
  float iv = sigmoidf_(x[d] + g[d]);
  float ov = sigmoidf_(x[H + d] + g[H + d]);
  float uv = tanhf(x[2 * H + d] + g[2 * H + d]);
  float cn = iv * uv + fc[idx];
  float hv = ov * tanhf(cn);
  c[(size_t)rw * H + d]  = cn;
  ht[(size_t)rw * H + d] = hv;
  htb[(size_t)rw * H + d] = (__bf16)hv;
}

// out = stack([c[root_ids], h[root_ids]])  -> [2, PER, H] f32
__global__ void k_out_gather(const float* __restrict__ c, const float* __restrict__ ht,
                             const int* __restrict__ root_ids, float* __restrict__ out) {
  int idx = blockIdx.x * blockDim.x + threadIdx.x;
  if (idx >= PER * H) return;
  int t = idx / H, d = idx % H;
  int rid = root_ids[t];
  out[idx]           = c[(size_t)rid * H + d];
  out[PER * H + idx] = ht[(size_t)rid * H + d];
}

// ---------------------------------------------------------------------------
// Host orchestration
// ---------------------------------------------------------------------------

static inline int nblk(long n) { return (int)((n + TB - 1) / TB); }

extern "C" void kernel_launch(void* const* d_in, const int* in_sizes, int n_in,
                              void* d_out, int out_size, void* d_ws, size_t ws_size,
                              hipStream_t stream) {
  (void)in_sizes; (void)n_in; (void)out_size; (void)ws_size;
  const float* x_seq = (const float*)d_in[0];   // [N, L, I]
  const float* W_ih  = (const float*)d_in[1];   // [I, 3H]
  const float* W_hh  = (const float*)d_in[2];   // [H, 3H]
  const float* b_ih  = (const float*)d_in[3];   // [3H]
  const float* b_hh  = (const float*)d_in[4];   // [3H]
  const float* Wx    = (const float*)d_in[5];   // [H, 4H]
  const float* bx    = (const float*)d_in[6];   // [4H]
  const float* U_iou = (const float*)d_in[7];   // [H, 3H]
  const float* U_f   = (const float*)d_in[8];   // [H, H]
  const int*   parent   = (const int*)d_in[9];  // [N]
  const int*   root_ids = (const int*)d_in[11]; // [PER]
  float* out = (float*)d_out;                   // [2, PER, H]

  // Workspace carve (256B aligned). GRU scratch is overlaid by tree scratch.
  char* base = (char*)d_ws;
  size_t off = 0;
  auto alloc = [&](size_t bytes) -> void* {
    void* r = base + off;
    off += (bytes + 255) & ~(size_t)255;
    return r;
  };
  __bf16* xb    = (__bf16*)alloc((size_t)N * L * I * 2);  // x in bf16
  __bf16* WihT  = (__bf16*)alloc((size_t)G3 * I * 2);     // [3H, I]
  __bf16* WhhT  = (__bf16*)alloc((size_t)G3 * H * 2);     // [3H, H]
  __bf16* WxT   = (__bf16*)alloc((size_t)G4 * H * 2);     // [4H, H]
  __bf16* UiouT = (__bf16*)alloc((size_t)G3 * H * 2);     // [3H, H]
  __bf16* UfT   = (__bf16*)alloc((size_t)H * H * 2);      // [H, H]
  float*  h     = (float*)alloc((size_t)N * H * 4);       // GRU h, then tree h
  __bf16* hb    = (__bf16*)alloc((size_t)N * H * 2);      // bf16 mirror of h
  float*  c     = (float*)alloc((size_t)N * H * 4);       // tree cell state
  const size_t scratch0 = off;
  float* gi = (float*)alloc((size_t)N * G3 * 4);          // GRU x-gates
  float* gh = (float*)alloc((size_t)N * G3 * 4);          // GRU h-gates
  off = scratch0;                                         // overlay for tree
  float*  xg    = (float*)alloc((size_t)N * G4 * 4);      // enc @ Wx + bx
  float*  giou  = (float*)alloc((size_t)PER * G3 * 4);
  float*  hUf   = (float*)alloc((size_t)PER * H * 4);
  float*  hsum  = (float*)alloc((size_t)PER * H * 4);
  __bf16* hsumb = (__bf16*)alloc((size_t)PER * H * 2);
  float*  fc    = (float*)alloc((size_t)PER * H * 4);

  // --- convert inputs to bf16 (weights transposed for the WMMA B path) ---
  k_cvt_bf16<<<nblk((long)N * L * I), TB, 0, stream>>>(x_seq, xb, N * L * I);
  k_cvt_transpose<<<nblk((long)I * G3), TB, 0, stream>>>(W_ih, WihT, I, G3);
  k_cvt_transpose<<<nblk((long)H * G3), TB, 0, stream>>>(W_hh, WhhT, H, G3);
  k_cvt_transpose<<<nblk((long)H * G4), TB, 0, stream>>>(Wx, WxT, H, G4);
  k_cvt_transpose<<<nblk((long)H * G3), TB, 0, stream>>>(U_iou, UiouT, H, G3);
  k_cvt_transpose<<<nblk((long)H * H), TB, 0, stream>>>(U_f, UfT, H, H);

  // --- GRU over L timesteps (sequential dependence through h) ---
  k_zero_f32<<<nblk((long)N * H), TB, 0, stream>>>(h, N * H);
  k_zero_bf16<<<nblk((long)N * H), TB, 0, stream>>>(hb, N * H);
  for (int t = 0; t < L; ++t) {
    // gi = x_t @ W_ih + b_ih   (x rows strided by L*I)
    k_gemm_bf16<I, L * I, G3><<<dim3(G3 / 128, N / 128), TB, 0, stream>>>(
        xb + (size_t)t * I, WihT, b_ih, gi);
    // gh = h @ W_hh + b_hh
    k_gemm_bf16<H, H, G3><<<dim3(G3 / 128, N / 128), TB, 0, stream>>>(
        hb, WhhT, b_hh, gh);
    k_gru_gate<<<nblk((long)N * H), TB, 0, stream>>>(gi, gh, h, hb);
  }

  // --- xg = enc @ Wx + bx ---
  k_gemm_bf16<H, H, G4><<<dim3(G4 / 128, N / 128), TB, 0, stream>>>(hb, WxT, bx, xg);

  // --- TreeLSTM: reuse h/hb as tree hidden state, zeroed ---
  k_zero_f32<<<nblk((long)N * H), TB, 0, stream>>>(h, N * H);
  k_zero_bf16<<<nblk((long)N * H), TB, 0, stream>>>(hb, N * H);
  k_zero_f32<<<nblk((long)N * H), TB, 0, stream>>>(c, N * H);

  for (int lev = D - 1; lev >= 0; --lev) {
    const int pbase = lev * PER;        // rows updated this level
    const int cbase = (lev + 1) * PER;  // child rows (if any)
    k_zero_f32<<<nblk((long)PER * H), TB, 0, stream>>>(hsum, PER * H);
    k_zero_f32<<<nblk((long)PER * H), TB, 0, stream>>>(fc, PER * H);
    if (lev < D - 1) {
      k_hsum_scatter<<<nblk((long)PER * H), TB, 0, stream>>>(h, parent, cbase, pbase, hsum);
    }
    k_cvt_bf16<<<nblk((long)PER * H), TB, 0, stream>>>(hsum, hsumb, PER * H);
    // giou = h_sum @ U_iou   (only the active level's 1024 rows)
    k_gemm_bf16<H, H, G3><<<dim3(G3 / 128, PER / 128), TB, 0, stream>>>(
        hsumb, UiouT, nullptr, giou);
    if (lev < D - 1) {
      // hUf = h_child @ U_f for the 1024 child rows, then fc scatter
      k_gemm_bf16<H, H, H><<<dim3(H / 128, PER / 128), TB, 0, stream>>>(
          hb + (size_t)cbase * H, UfT, nullptr, hUf);
      k_fc_scatter<<<nblk((long)PER * H), TB, 0, stream>>>(xg, hUf, c, parent, cbase, pbase, fc);
    }
    k_level_update<<<nblk((long)PER * H), TB, 0, stream>>>(xg, giou, fc, pbase, c, h, hb);
  }

  // --- gather root (c, h) ---
  k_out_gather<<<nblk((long)PER * H), TB, 0, stream>>>(c, h, root_ids, out);
}
